// Attention_14663018349197
// MI455X (gfx1250) — matrix operand order repaired and checked by reading, not compiled, not run on hardware
//
#include <hip/hip_runtime.h>

// ---------------------------------------------------------------------------
// Problem constants (from reference): B=1024, L=86, D=512, DV=64, NV=8
// Flattened token count BL = 88064; tile 64 tokens per workgroup -> 1376 WGs.
// ---------------------------------------------------------------------------
constexpr int DIM       = 512;
constexpr int TILE_M    = 64;
constexpr int LDS_STR   = 520;          // 512 + 8 halves pad (bank spread)
constexpr int NBLOCKS   = (1024 * 86) / TILE_M;   // 1376
constexpr int WELEM     = DIM * DIM;    // 262144 elements per weight matrix

typedef __attribute__((ext_vector_type(16))) __bf16 v16bf;
typedef __attribute__((ext_vector_type(8)))  __bf16 v8bf;
typedef __attribute__((ext_vector_type(8)))  float  v8f;
typedef __attribute__((ext_vector_type(4)))  float  v4f;

// Force global (AS1) memory ops so weight streams use global_load_b128
// (LOADcnt only) instead of flat_load (LOADcnt+DScnt, couples with LDS waits).
#define GAS __attribute__((address_space(1)))
__device__ __forceinline__ const GAS v8bf* gvp(const __bf16* p) {
    return (const GAS v8bf*)(uintptr_t)p;
}
__device__ __forceinline__ const GAS v4f* gfp4(const float* p) {
    return (const GAS v4f*)(uintptr_t)p;
}
__device__ __forceinline__ const GAS float* gfp(const float* p) {
    return (const GAS float*)(uintptr_t)p;
}
__device__ __forceinline__ GAS float* gfp_mut(float* p) {
    return (GAS float*)(uintptr_t)p;
}

// ---------------------------------------------------------------------------
// Kernel 0: convert fp32 weights [K][N] -> bf16 transposed [N][K] in d_ws.
// Four matrices: Wq, Wk, Wv, Wfc.
// ---------------------------------------------------------------------------
__global__ __launch_bounds__(256) void prep_weights(
    const float* __restrict__ Wq, const float* __restrict__ Wk,
    const float* __restrict__ Wv, const float* __restrict__ Wfc,
    __bf16* __restrict__ ws)
{
    int idx = blockIdx.x * 256 + threadIdx.x;          // 0 .. 4*262144-1
    int mtx = idx >> 18;
    int e   = idx & (WELEM - 1);
    int n   = e >> 9;          // output row  (N)
    int k   = e & 511;         // output col  (K)
    const float* W = (mtx == 0) ? Wq : (mtx == 1) ? Wk : (mtx == 2) ? Wv : Wfc;
    ws[(size_t)mtx * WELEM + e] = (__bf16)W[k * DIM + n];
}

// ---------------------------------------------------------------------------
// Stage a 64x512 fp32 tile from global into LDS as bf16 (row stride 520).
// ---------------------------------------------------------------------------
__device__ inline void stage_x(const float* __restrict__ X, __bf16* xlds,
                               int m0, int tid)
{
    for (int i = tid * 8; i < TILE_M * DIM; i += 256 * 8) {
        int r = i >> 9;
        int c = i & 511;
        const v4f a = *gfp4(X + (size_t)(m0 + r) * DIM + c);
        const v4f b = *gfp4(X + (size_t)(m0 + r) * DIM + c + 4);
        v8bf o;
        o[0] = (__bf16)a.x; o[1] = (__bf16)a.y; o[2] = (__bf16)a.z; o[3] = (__bf16)a.w;
        o[4] = (__bf16)b.x; o[5] = (__bf16)b.y; o[6] = (__bf16)b.z; o[7] = (__bf16)b.w;
        *(v8bf*)(xlds + r * LDS_STR + c) = o;
    }
}

// A fragment: 16x32 bf16 tile from LDS (row-major, stride LDS_STR).
// Per ISA: lane<16 -> M=lane, K = {0..7, 16..23}; lane>=16 -> K = {8..15, 24..31}.
__device__ inline v16bf load_a_frag(const __bf16* lds, int rowBase, int k0, int lane)
{
    int s = (lane < 16) ? 0 : 8;
    int m = rowBase + (lane & 15);
    const __bf16* p = lds + m * LDS_STR + k0 + s;
    v8bf lo = *(const v8bf*)(p);
    v8bf hi = *(const v8bf*)(p + 16);
    v16bf a;
#pragma unroll
    for (int i = 0; i < 8; ++i) { a[i] = lo[i]; a[i + 8] = hi[i]; }
    return a;
}

// B fragment: 32x16 bf16 tile from transposed weights WT[N][K] (global).
// Per ISA (column-major striping): lane<16 -> N=lane, K = k0+{0..7, 16..23};
// lane>=16 -> N=lane-16, K = k0+{8..15, 24..31}.
__device__ inline v16bf load_b_frag(const __bf16* __restrict__ WT,
                                    int n0, int k0, int lane)
{
    int t = (lane < 16) ? 0 : 8;
    const __bf16* p = WT + (size_t)(n0 + (lane & 15)) * DIM + k0 + t;
    v8bf lo = *gvp(p);
    v8bf hi = *gvp(p + 16);
    v16bf b;
#pragma unroll
    for (int i = 0; i < 8; ++i) { b[i] = lo[i]; b[i + 8] = hi[i]; }
    return b;
}

// ---------------------------------------------------------------------------
// 64x512 = relu(64x512 @ 512x512 + bias) GEMM with bf16 WMMA, fp32 accum.
// Each wave owns a 64(M) x 64(N) sub-tile: 4x4 grid of 16x16 accumulators.
// Epilogue: +bias, ReLU, then either bf16 -> LDS or fp32 -> global.
// ---------------------------------------------------------------------------
template <bool TO_GLOBAL>
__device__ void gemm_512(const __bf16* Alds, const __bf16* __restrict__ WT,
                         const float* __restrict__ bias,
                         __bf16* dstLds, float* __restrict__ dstGlobal,
                         int m0, int tid)
{
    const int lane = tid & 31;
    const int wave = tid >> 5;
    const int n0   = wave * 64;        // 8 waves cover N = 512

    v8f acc[4][4];
#pragma unroll
    for (int mt = 0; mt < 4; ++mt)
#pragma unroll
        for (int nt = 0; nt < 4; ++nt)
            acc[mt][nt] = (v8f)(0.0f);

    for (int k0 = 0; k0 < DIM; k0 += 32) {
        v16bf a[4], b[4];
#pragma unroll
        for (int mt = 0; mt < 4; ++mt) a[mt] = load_a_frag(Alds, mt * 16, k0, lane);
#pragma unroll
        for (int nt = 0; nt < 4; ++nt) b[nt] = load_b_frag(WT, n0 + nt * 16, k0, lane);
#pragma unroll
        for (int mt = 0; mt < 4; ++mt)
#pragma unroll
            for (int nt = 0; nt < 4; ++nt)
                acc[mt][nt] = __builtin_amdgcn_wmma_f32_16x16x32_bf16(
                    false, a[mt], false, b[nt], (short)0, acc[mt][nt],
                    false, false);
    }

    // Epilogue.  C/D layout: VGPR r -> M = r (+8 for lanes>=16); lane%16 -> N.
#pragma unroll
    for (int nt = 0; nt < 4; ++nt) {
        const int n  = n0 + nt * 16 + (lane & 15);
        const float bv = *gfp(bias + n);
#pragma unroll
        for (int mt = 0; mt < 4; ++mt) {
            const int mb = mt * 16 + ((lane < 16) ? 0 : 8);
#pragma unroll
            for (int r = 0; r < 8; ++r) {
                float v = acc[mt][nt][r] + bv;
                v = v > 0.0f ? v : 0.0f;
                if (TO_GLOBAL)
                    *gfp_mut(dstGlobal + (size_t)(m0 + mb + r) * DIM + n) = v;
                else
                    dstLds[(mb + r) * LDS_STR + n] = (__bf16)v;
            }
        }
    }
}

// ---------------------------------------------------------------------------
// Cross-head attention per token (8 heads x 64 dims), softmax over 8 heads,
// ctx + residual(query) -> bf16 into outLds.  2 (token,head) rows per thread.
// ---------------------------------------------------------------------------
__device__ void attention(const __bf16* q2, const __bf16* k2, const __bf16* v2,
                          __bf16* outLds, const float* __restrict__ query,
                          int m0, int tid)
{
    const int t     = tid & 63;
    const int nbase = tid >> 6;           // 0..3
#pragma unroll 1
    for (int half = 0; half < 2; ++half) {
        const int n = nbase + half * 4;   // head index 0..7
        // --- load q row (64 fp32) ---
        float qr[64];
        const __bf16* qp = q2 + t * LDS_STR + n * 64;
#pragma unroll
        for (int c = 0; c < 8; ++c) {
            v8bf v = *(const v8bf*)(qp + 8 * c);
#pragma unroll
            for (int j = 0; j < 8; ++j) qr[8 * c + j] = (float)v[j];
        }
        // --- scores over 8 heads ---
        float s[8];
#pragma unroll
        for (int m = 0; m < 8; ++m) {
            const __bf16* kp = k2 + t * LDS_STR + m * 64;
            float d = 0.0f;
#pragma unroll
            for (int c = 0; c < 8; ++c) {
                v8bf v = *(const v8bf*)(kp + 8 * c);
#pragma unroll
                for (int j = 0; j < 8; ++j) d += qr[8 * c + j] * (float)v[j];
            }
            s[m] = d * 0.125f;            // 1/sqrt(64)
        }
        // --- softmax ---
        float mx = s[0];
#pragma unroll
        for (int m = 1; m < 8; ++m) mx = s[m] > mx ? s[m] : mx;
        float sum = 0.0f;
#pragma unroll
        for (int m = 0; m < 8; ++m) { s[m] = __expf(s[m] - mx); sum += s[m]; }
        const float inv = 1.0f / sum;
        // --- ctx = p @ v2 ---
        float ctx[64];
#pragma unroll
        for (int j = 0; j < 64; ++j) ctx[j] = 0.0f;
#pragma unroll
        for (int m = 0; m < 8; ++m) {
            const float p = s[m] * inv;
            const __bf16* vp = v2 + t * LDS_STR + m * 64;
#pragma unroll
            for (int c = 0; c < 8; ++c) {
                v8bf v = *(const v8bf*)(vp + 8 * c);
#pragma unroll
                for (int j = 0; j < 8; ++j) ctx[8 * c + j] += p * (float)v[j];
            }
        }
        // --- residual (vectorized) + store bf16 row into outLds (16B stores) ---
        const float* qg = query + (size_t)(m0 + t) * DIM + n * 64;
        __bf16* op = outLds + t * LDS_STR + n * 64;
#pragma unroll
        for (int c = 0; c < 8; ++c) {
            const v4f ra = *gfp4(qg + 8 * c);
            const v4f rb = *gfp4(qg + 8 * c + 4);
            v8bf o;
            o[0] = (__bf16)(ctx[8 * c + 0] + ra.x);
            o[1] = (__bf16)(ctx[8 * c + 1] + ra.y);
            o[2] = (__bf16)(ctx[8 * c + 2] + ra.z);
            o[3] = (__bf16)(ctx[8 * c + 3] + ra.w);
            o[4] = (__bf16)(ctx[8 * c + 4] + rb.x);
            o[5] = (__bf16)(ctx[8 * c + 5] + rb.y);
            o[6] = (__bf16)(ctx[8 * c + 6] + rb.z);
            o[7] = (__bf16)(ctx[8 * c + 7] + rb.w);
            *(v8bf*)(op + 8 * c) = o;
        }
    }
}

// ---------------------------------------------------------------------------
// Fused kernel: QKV projections -> cross-head attention -> residual -> FC.
// One workgroup = 64 tokens; dynamic LDS = 4 * 64*520 bf16 tiles (~260 KB).
// ---------------------------------------------------------------------------
extern "C" __global__ __launch_bounds__(256) void fused_attn_kernel(
    const float* __restrict__ query, const float* __restrict__ key,
    const float* __restrict__ value,
    const float* __restrict__ bq, const float* __restrict__ bk,
    const float* __restrict__ bv, const float* __restrict__ bfc,
    const __bf16* __restrict__ WqT, const __bf16* __restrict__ WkT,
    const __bf16* __restrict__ WvT, const __bf16* __restrict__ WfcT,
    float* __restrict__ out)
{
    extern __shared__ __bf16 smem[];
    __bf16* q2 = smem;                       // 64 x 520
    __bf16* k2 = q2 + TILE_M * LDS_STR;
    __bf16* v2 = k2 + TILE_M * LDS_STR;
    __bf16* xs = v2 + TILE_M * LDS_STR;      // staging / ctx tile

    const int tid = threadIdx.x;
    const int m0  = blockIdx.x * TILE_M;

    // Q projection
    stage_x(query, xs, m0, tid);
    __syncthreads();
    gemm_512<false>(xs, WqT, bq, q2, nullptr, m0, tid);
    __syncthreads();
    // K projection
    stage_x(key, xs, m0, tid);
    __syncthreads();
    gemm_512<false>(xs, WkT, bk, k2, nullptr, m0, tid);
    __syncthreads();
    // V projection
    stage_x(value, xs, m0, tid);
    __syncthreads();
    gemm_512<false>(xs, WvT, bv, v2, nullptr, m0, tid);
    __syncthreads();
    // Cross-head attention + residual -> xs (bf16)
    attention(q2, k2, v2, xs, query, m0, tid);
    __syncthreads();
    // FC + ReLU -> global fp32
    gemm_512<true>(xs, WfcT, bfc, nullptr, out, m0, tid);
}

// ---------------------------------------------------------------------------
// Host entry
// ---------------------------------------------------------------------------
extern "C" void kernel_launch(void* const* d_in, const int* in_sizes, int n_in,
                              void* d_out, int out_size, void* d_ws, size_t ws_size,
                              hipStream_t stream)
{
    const float* query = (const float*)d_in[0];
    const float* key   = (const float*)d_in[1];
    const float* value = (const float*)d_in[2];
    const float* Wq    = (const float*)d_in[3];
    const float* bq    = (const float*)d_in[4];
    const float* Wk    = (const float*)d_in[5];
    const float* bk    = (const float*)d_in[6];
    const float* Wv    = (const float*)d_in[7];
    const float* bv    = (const float*)d_in[8];
    const float* Wfc   = (const float*)d_in[9];
    const float* bfc   = (const float*)d_in[10];

    __bf16* WT = (__bf16*)d_ws;   // 4 transposed bf16 weight matrices (2 MB)

    prep_weights<<<(4 * WELEM) / 256, 256, 0, stream>>>(Wq, Wk, Wv, Wfc, WT);

    const size_t ldsBytes = (size_t)4 * TILE_M * LDS_STR * sizeof(__bf16); // 266240
    (void)hipFuncSetAttribute((const void*)fused_attn_kernel,
                              hipFuncAttributeMaxDynamicSharedMemorySize,
                              (int)ldsBytes);

    fused_attn_kernel<<<NBLOCKS, 256, ldsBytes, stream>>>(
        query, key, value, bq, bk, bv, bfc,
        WT, WT + WELEM, WT + 2 * WELEM, WT + 3 * WELEM,
        (float*)d_out);
}
